// T5Attention_16501264351489
// MI455X (gfx1250) — matrix operand order repaired and checked by reading, not compiled, not run on hardware
//
#include <hip/hip_runtime.h>

// ---------------- problem constants ----------------
#define BATCH   2
#define SEQ     2048
#define DMODEL  1024
#define HD      16
#define DH      64
#define NBUCK   32
#define TABW    (2*SEQ - 1)   // 4095 relative positions

typedef __attribute__((ext_vector_type(16))) __bf16 v16bf;
typedef __attribute__((ext_vector_type(8)))  __bf16 v8bf;
typedef __attribute__((ext_vector_type(8)))  float  v8f;
typedef __attribute__((ext_vector_type(4)))  unsigned int u32x4;
typedef __attribute__((ext_vector_type(8)))  int          i32x8;
typedef __attribute__((ext_vector_type(4)))  int          i32x4;

union V16U {
    v16bf v;
    struct Halves { v8bf lo, hi; } p;
};

__device__ __forceinline__ v8f splat8(float x) {
    v8f r;
#pragma unroll
    for (int i = 0; i < 8; ++i) r[i] = x;
    return r;
}

__device__ __forceinline__ v8f vmax8(v8f a, v8f b) {
    v8f r;
#pragma unroll
    for (int i = 0; i < 8; ++i) r[i] = fmaxf(a[i], b[i]);
    return r;
}

__device__ __forceinline__ v8f wmma_bf16(v16bf a, v16bf b, v8f c) {
    return __builtin_amdgcn_wmma_f32_16x16x32_bf16(
        false, a, false, b, (short)0, c, false, false);
}

// Raw LDS byte offset: flat addressing truncates generic addr to [31:0] for LDS.
__device__ __forceinline__ unsigned lds_addr_of(const void* p) {
    return (unsigned)(unsigned long long)p;
}

// Async global->LDS copy (16B per lane), tracked by ASYNCcnt.
__device__ __forceinline__ void async_load_b128(unsigned lds, const void* g) {
    asm volatile("global_load_async_to_lds_b128 %0, %1, off"
                 :: "v"(lds), "v"((unsigned long long)g) : "memory");
}
__device__ __forceinline__ void wait_async0() {
    asm volatile("s_wait_asynccnt 0x0" ::: "memory");
}

// ---------------- convert f32 -> bf16 ----------------
__global__ void k_f32_to_bf16(const float* __restrict__ src,
                              __bf16* __restrict__ dst, int n) {
    int i = blockIdx.x * blockDim.x + threadIdx.x;
    if (i < n) dst[i] = (__bf16)src[i];
}

// ---------------- relative-position bias table ----------------
__global__ void k_bias_table(const float* __restrict__ rel_emb,
                             float* __restrict__ tab) {
    int i = blockIdx.x * blockDim.x + threadIdx.x;
    if (i >= HD * TABW) return;
    int h = i / TABW;
    int r = i % TABW - (SEQ - 1);          // rel_pos = k - q
    const int nb = NBUCK / 2;              // 16
    int bucket = (r > 0) ? nb : 0;
    int rel = (r < 0) ? -r : r;
    const int max_exact = nb / 2;          // 8
    int bv;
    if (rel < max_exact) {
        bv = rel;
    } else {
        float t = __logf((float)rel / (float)max_exact) /
                  __logf(128.0f / (float)max_exact) * (float)(nb - max_exact);
        bv = max_exact + (int)t;
        if (bv > nb - 1) bv = nb - 1;
    }
    bucket += bv;
    tab[i] = rel_emb[bucket * HD + h];
}

// ---------------- bf16 GEMM: C[M,N] = A[M,K] * B[K,N] ----------------
// 128x128x32 tiles, 256 threads = 8 waves, each wave: 2(M) x 4(N) WMMA subtiles.
// A tile staged with ASYNC global->LDS loads; next tiles prefetched into GL2.
template <bool OUT_BF16>
__global__ __launch_bounds__(256)
void k_gemm_bf16(const __bf16* __restrict__ A, const __bf16* __restrict__ B,
                 void* __restrict__ Cout, int M, int N, int K) {
    __shared__ __attribute__((aligned(16))) __bf16 At[128][40];  // [m][k], 80B rows
    __shared__ __attribute__((aligned(16))) __bf16 Bt[128][40];  // [n][k] (transposed)

    const int tid  = threadIdx.x;
    const int wave = tid >> 5;
    const int lane = tid & 31;
    const int l16  = lane & 15;
    const int half = lane >> 4;

    const int row0  = blockIdx.y * 128;
    const int col0  = blockIdx.x * 128;
    const int waveM = (wave >> 1) * 32;
    const int waveN = (wave & 1) * 64;

    v8f acc[2][4];
#pragma unroll
    for (int i = 0; i < 2; ++i)
#pragma unroll
        for (int j = 0; j < 4; ++j) acc[i][j] = splat8(0.0f);

    for (int k0 = 0; k0 < K; k0 += 32) {
        // A tile 128x32: async copy straight into LDS (16B per lane)
#pragma unroll
        for (int p = 0; p < 2; ++p) {
            int c  = tid + p * 256;          // 512 chunks of 8 elems
            int r  = c >> 2;
            int kc = (c & 3) * 8;
            const __bf16* src = A + (size_t)(row0 + r) * K + k0 + kc;
            async_load_b128(lds_addr_of(&At[r][kc]), src);
            __builtin_prefetch(src + 32, 0, 1);   // next k-tile -> GL2
        }
        // B tile 32x128, transpose into Bt[n][k]
#pragma unroll
        for (int p = 0; p < 2; ++p) {
            int c  = tid + p * 256;
            int kr = c >> 4;
            int nc = (c & 15) * 8;
            const __bf16* src = B + (size_t)(k0 + kr) * N + col0 + nc;
            v8bf b = *(const v8bf*)src;
            __builtin_prefetch(src + (size_t)32 * N, 0, 1);
#pragma unroll
            for (int e = 0; e < 8; ++e) Bt[nc + e][kr] = b[e];
        }
        wait_async0();
        __syncthreads();

        V16U afrag[2], bfrag[4];
#pragma unroll
        for (int ms = 0; ms < 2; ++ms) {
            int r = waveM + ms * 16 + l16;
            afrag[ms].p.lo = *(const v8bf*)&At[r][half * 8];        // K 0-7 / 8-15
            afrag[ms].p.hi = *(const v8bf*)&At[r][16 + half * 8];   // K 16-23 / 24-31
        }
#pragma unroll
        for (int ns = 0; ns < 4; ++ns) {
            int n = waveN + ns * 16 + l16;
            bfrag[ns].p.lo = *(const v8bf*)&Bt[n][half * 8];        // K 0-7 / 8-15
            bfrag[ns].p.hi = *(const v8bf*)&Bt[n][16 + half * 8];   // K 16-23 / 24-31
        }
#pragma unroll
        for (int ms = 0; ms < 2; ++ms)
#pragma unroll
            for (int ns = 0; ns < 4; ++ns)
                acc[ms][ns] = wmma_bf16(afrag[ms].v, bfrag[ns].v, acc[ms][ns]);
        __syncthreads();
    }

#pragma unroll
    for (int ms = 0; ms < 2; ++ms)
#pragma unroll
        for (int ns = 0; ns < 4; ++ns)
#pragma unroll
            for (int j = 0; j < 8; ++j) {
                int r  = row0 + waveM + ms * 16 + j + 8 * half;
                int cN = col0 + waveN + ns * 16 + l16;
                if (OUT_BF16)
                    ((__bf16*)Cout)[(size_t)r * N + cN] = (__bf16)acc[ms][ns][j];
                else
                    ((float*)Cout)[(size_t)r * N + cN] = acc[ms][ns][j];
            }
}

// ---------------- flash attention per (b, h, 64-row q-tile) ----------------
// 128 threads = 4 waves. K tile staged by the Tensor Data Mover (TDM) with
// hardware LDS padding (32-DWORD interval, 4-DWORD pad -> 72-elem rows).
__global__ __launch_bounds__(128)
void k_attention(const __bf16* __restrict__ Q, const __bf16* __restrict__ Kk,
                 const __bf16* __restrict__ V, const float* __restrict__ btab,
                 __bf16* __restrict__ O) {
    __shared__ __attribute__((aligned(16))) __bf16 Kt[64][72];      // [kcol][d]
    __shared__ __attribute__((aligned(16))) __bf16 Vt[64][72];      // [d][kcol]
    __shared__ __attribute__((aligned(16))) __bf16 Pw[4][16][72];   // per-wave P

    const int tid  = threadIdx.x;
    const int wave = tid >> 5;
    const int lane = tid & 31;
    const int l16  = lane & 15;
    const int half = lane >> 4;

    const int qt = blockIdx.x & 31;
    const int h  = (blockIdx.x >> 5) & 15;
    const int b  = blockIdx.x >> 9;

    const int    q0        = qt * 64 + wave * 16;
    const size_t rowstride = HD * DH;                         // 1024
    const size_t base      = (size_t)b * SEQ * rowstride + (size_t)h * DH;

    const unsigned KtLds = lds_addr_of(&Kt[0][0]);

    // Q fragments (A layout), kept in registers
    V16U qa[2];
    {
        const __bf16* qrow = Q + base + (size_t)(q0 + l16) * rowstride;
        qa[0].p.lo = *(const v8bf*)(qrow + half * 8);
        qa[0].p.hi = *(const v8bf*)(qrow + 16 + half * 8);
        qa[1].p.lo = *(const v8bf*)(qrow + 32 + half * 8);
        qa[1].p.hi = *(const v8bf*)(qrow + 48 + half * 8);
    }

    v8f o[4];
#pragma unroll
    for (int nt = 0; nt < 4; ++nt) o[nt] = splat8(0.0f);
    v8f m = splat8(-1e30f);
    v8f l = splat8(0.0f);

    for (int kt = 0; kt < SEQ / 64; ++kt) {
        // ---- K tile via TDM: 64x64 bf16 tile, row stride 1024 elems ----
        if (wave == 0) {
            unsigned long long ga =
                (unsigned long long)(Kk + base + (size_t)(kt * 64) * rowstride);
            u32x4 g0;
            g0[0] = 1u;                                   // count=1 (valid D#)
            g0[1] = KtLds;                                // lds_addr
            g0[2] = (unsigned)(ga & 0xffffffffull);       // global_addr[31:0]
            g0[3] = (unsigned)((ga >> 32) & 0x01ffffffu)  // global_addr[56:32]
                    | (2u << 30);                         // type = 2 (image)
            i32x8 g1;
            g1[0] = (1 << 16)      // data_size = 2B
                  | (1 << 20)      // pad_enable
                  | (4 << 22)      // pad_interval: 2^(4+1)=32 DWORDs (64 elems)
                  | (3 << 25);     // pad_amount: 3+1 = 4 DWORDs (8 elems)
            g1[1] = (int)(64u << 16);     // tensor_dim0 = 64 (lo16 in [31:16])
            g1[2] = (int)(2048u << 16);   // tensor_dim1 = 2048 (lo16 in [31:16])
            g1[3] = (int)(64u << 16);     // tile_dim0 = 64
            g1[4] = 64;                   // tile_dim1 = 64 (tile_dim2 = 0)
            g1[5] = 1024;                 // tensor_dim0_stride = 1024 elems
            g1[6] = 0;
            g1[7] = 0;
            i32x4 z4 = {0, 0, 0, 0};
            i32x8 z8 = {0, 0, 0, 0, 0, 0, 0, 0};
            __builtin_amdgcn_tensor_load_to_lds(g0, g1, z4, z4, z8, 0);
        }

        // ---- V tile: manual transposed store Vt[d][kcol] ----
#pragma unroll
        for (int p = 0; p < 4; ++p) {
            int c  = tid + p * 128;
            int r  = c >> 3;                  // kcol 0..63
            int cc = (c & 7) * 8;             // d offset
            const __bf16* vs = V + base + (size_t)(kt * 64 + r) * rowstride + cc;
            v8bf vv = *(const v8bf*)vs;
            __builtin_prefetch(vs + (size_t)64 * rowstride, 0, 1);  // next tile
#pragma unroll
            for (int e = 0; e < 8; ++e) Vt[cc + e][r] = vv[e];
        }
        if (wave == 0) __builtin_amdgcn_s_wait_tensorcnt(0);
        __syncthreads();

        // ---- scores S = Q * K^T (+ rel-pos bias) ----
        v8f s[4];
#pragma unroll
        for (int nt = 0; nt < 4; ++nt) {
            V16U b0, b1;
            int n = nt * 16 + l16;
            b0.p.lo = *(const v8bf*)&Kt[n][half * 8];
            b0.p.hi = *(const v8bf*)&Kt[n][16 + half * 8];
            b1.p.lo = *(const v8bf*)&Kt[n][32 + half * 8];
            b1.p.hi = *(const v8bf*)&Kt[n][48 + half * 8];
            v8f z = splat8(0.0f);
            z = wmma_bf16(qa[0].v, b0.v, z);
            z = wmma_bf16(qa[1].v, b1.v, z);
            int kcol = kt * 64 + nt * 16 + l16;
            const float* tb = btab + (size_t)h * TABW + (SEQ - 1)
                              + kcol - (q0 + 8 * half);
#pragma unroll
            for (int j = 0; j < 8; ++j) z[j] += tb[-j];
            s[nt] = z;
        }

        // ---- online softmax (rows replicated across each 16-lane half) ----
        v8f tmax = vmax8(vmax8(s[0], s[1]), vmax8(s[2], s[3]));
#pragma unroll
        for (int mask = 1; mask < 16; mask <<= 1)
#pragma unroll
            for (int j = 0; j < 8; ++j)
                tmax[j] = fmaxf(tmax[j], __shfl_xor(tmax[j], mask, 16));
        v8f mnew = vmax8(m, tmax);
        v8f alpha;
#pragma unroll
        for (int j = 0; j < 8; ++j) alpha[j] = __expf(m[j] - mnew[j]);
        m = mnew;

        v8f rsum = splat8(0.0f);
#pragma unroll
        for (int nt = 0; nt < 4; ++nt)
#pragma unroll
            for (int j = 0; j < 8; ++j) {
                float pv = __expf(s[nt][j] - mnew[j]);
                s[nt][j] = pv;
                rsum[j] += pv;
            }
#pragma unroll
        for (int mask = 1; mask < 16; mask <<= 1)
#pragma unroll
            for (int j = 0; j < 8; ++j)
                rsum[j] += __shfl_xor(rsum[j], mask, 16);
#pragma unroll
        for (int j = 0; j < 8; ++j) l[j] = l[j] * alpha[j] + rsum[j];
#pragma unroll
        for (int nt = 0; nt < 4; ++nt)
#pragma unroll
            for (int j = 0; j < 8; ++j) o[nt][j] *= alpha[j];

        // ---- C-layout -> A-layout via per-wave LDS (same-wave, in-order) ----
#pragma unroll
        for (int nt = 0; nt < 4; ++nt)
#pragma unroll
            for (int j = 0; j < 8; ++j)
                Pw[wave][j + 8 * half][nt * 16 + l16] = (__bf16)s[nt][j];

        V16U pa[2];
        {
            const __bf16* pr = &Pw[wave][l16][0];
            pa[0].p.lo = *(const v8bf*)(pr + half * 8);
            pa[0].p.hi = *(const v8bf*)(pr + 16 + half * 8);
            pa[1].p.lo = *(const v8bf*)(pr + 32 + half * 8);
            pa[1].p.hi = *(const v8bf*)(pr + 48 + half * 8);
        }
        // ---- O += P * V ----
#pragma unroll
        for (int nt = 0; nt < 4; ++nt) {
            V16U b0, b1;
            int d = nt * 16 + l16;
            b0.p.lo = *(const v8bf*)&Vt[d][half * 8];
            b0.p.hi = *(const v8bf*)&Vt[d][16 + half * 8];
            b1.p.lo = *(const v8bf*)&Vt[d][32 + half * 8];
            b1.p.hi = *(const v8bf*)&Vt[d][48 + half * 8];
            o[nt] = wmma_bf16(pa[0].v, b0.v, o[nt]);
            o[nt] = wmma_bf16(pa[1].v, b1.v, o[nt]);
        }
        __syncthreads();   // protect Kt/Vt before next k-tile
    }

    // normalize and write bf16 output [B*S, H*DH]
    v8f linv;
#pragma unroll
    for (int j = 0; j < 8; ++j) linv[j] = 1.0f / l[j];
#pragma unroll
    for (int nt = 0; nt < 4; ++nt)
#pragma unroll
        for (int j = 0; j < 8; ++j) {
            int r = q0 + j + 8 * half;
            O[base + (size_t)r * rowstride + nt * 16 + l16] =
                (__bf16)(o[nt][j] * linv[j]);
        }
}

// ---------------- host orchestration ----------------
extern "C" void kernel_launch(void* const* d_in, const int* in_sizes, int n_in,
                              void* d_out, int out_size, void* d_ws, size_t ws_size,
                              hipStream_t stream) {
    (void)in_sizes; (void)n_in; (void)out_size; (void)ws_size;
    const float* hs  = (const float*)d_in[0];
    const float* Wq  = (const float*)d_in[1];
    const float* Wk  = (const float*)d_in[2];
    const float* Wv  = (const float*)d_in[3];
    const float* Wo  = (const float*)d_in[4];
    const float* rel = (const float*)d_in[5];

    char*  ws  = (char*)d_ws;
    size_t off = 0;
    auto alloc = [&](size_t bytes) -> void* {
        off = (off + 255) & ~(size_t)255;
        void* p = ws + off;
        off += bytes;
        return p;
    };

    const size_t MROWS = (size_t)BATCH * SEQ;   // 4096
    __bf16* Xb   = (__bf16*)alloc(MROWS * DMODEL * 2);
    __bf16* Wqb  = (__bf16*)alloc((size_t)DMODEL * DMODEL * 2);
    __bf16* Wkb  = (__bf16*)alloc((size_t)DMODEL * DMODEL * 2);
    __bf16* Wvb  = (__bf16*)alloc((size_t)DMODEL * DMODEL * 2);
    __bf16* Wob  = (__bf16*)alloc((size_t)DMODEL * DMODEL * 2);
    __bf16* Qb   = (__bf16*)alloc(MROWS * DMODEL * 2);
    __bf16* Kb   = (__bf16*)alloc(MROWS * DMODEL * 2);
    __bf16* Vb   = (__bf16*)alloc(MROWS * DMODEL * 2);
    float*  btab = (float*)alloc((size_t)HD * TABW * 4);
    __bf16* AOb  = Xb;   // X no longer needed after the QKV projections

    const int nHS = (int)(MROWS * DMODEL);
    const int nW  = DMODEL * DMODEL;
    const int nT  = HD * TABW;
    k_f32_to_bf16<<<(nHS + 255) / 256, 256, 0, stream>>>(hs, Xb, nHS);
    k_f32_to_bf16<<<(nW + 255) / 256, 256, 0, stream>>>(Wq, Wqb, nW);
    k_f32_to_bf16<<<(nW + 255) / 256, 256, 0, stream>>>(Wk, Wkb, nW);
    k_f32_to_bf16<<<(nW + 255) / 256, 256, 0, stream>>>(Wv, Wvb, nW);
    k_f32_to_bf16<<<(nW + 255) / 256, 256, 0, stream>>>(Wo, Wob, nW);
    k_bias_table<<<(nT + 255) / 256, 256, 0, stream>>>(rel, btab);

    dim3 ggrid(DMODEL / 128, (unsigned)(MROWS / 128));   // (8, 32)
    k_gemm_bf16<true><<<ggrid, 256, 0, stream>>>(Xb, Wqb, (void*)Qb,
                                                 (int)MROWS, DMODEL, DMODEL);
    k_gemm_bf16<true><<<ggrid, 256, 0, stream>>>(Xb, Wkb, (void*)Kb,
                                                 (int)MROWS, DMODEL, DMODEL);
    k_gemm_bf16<true><<<ggrid, 256, 0, stream>>>(Xb, Wvb, (void*)Vb,
                                                 (int)MROWS, DMODEL, DMODEL);

    k_attention<<<BATCH * HD * (SEQ / 64), 128, 0, stream>>>(Qb, Kb, Vb, btab, AOb);

    k_gemm_bf16<false><<<ggrid, 256, 0, stream>>>(AOb, Wob, d_out,
                                                  (int)MROWS, DMODEL, DMODEL);
}
